// TriMul_13022340841597
// MI455X (gfx1250) — matrix operand order repaired and checked by reading, not compiled, not run on hardware
//
#include <hip/hip_runtime.h>

// ---------------------------------------------------------------------------
// TriMul for MI455X (gfx1250, wave32, WMMA bf16 16x16x32)
//   N = 768 (pair dim), D = 128 (in channels), H = 128 (hidden)
// Phase 1: LayerNorm + 5 projections (WMMA bf16) + gating -> bf16 workspace
//          leftA [d][i][k], rightR [d][j][k]  (k contiguous), og [m][d]
// Phase 2: per-channel GEMM C_d = A_d * R_d^T  (WMMA bf16, dbl-buffered LDS,
//          async global->LDS copies, ASYNCcnt-tracked) -> out_mid [d][i][j]
// Phase 3: LayerNorm over d + gate + final projection (WMMA bf16) -> out f32
// ---------------------------------------------------------------------------

#define NDIM 768
#define DDIM 128
#define HDIM 128
#define ROWS 64              // (i,k) positions handled per block in phase1/3
#define LDP  (DDIM + 8)      // padded LDS row stride in bf16 halves (=136)

// Async global->LDS (gfx1250 GLOBAL_LOAD_ASYNC_TO_LDS_*, ASYNCcnt-tracked).
#if defined(__has_builtin)
#if __has_builtin(__builtin_amdgcn_global_load_async_to_lds_b128) && \
    __has_builtin(__builtin_amdgcn_s_wait_asynccnt)
#define USE_ASYNC_LDS 1
#endif
#endif
#ifndef USE_ASYNC_LDS
#define USE_ASYNC_LDS 0
#endif

typedef __attribute__((ext_vector_type(4))) int v4i;
#if USE_ASYNC_LDS
typedef __attribute__((address_space(1))) v4i* gptr_t;  // global int4*
typedef __attribute__((address_space(3))) v4i* lptr_t;  // LDS int4*
#endif

typedef __attribute__((ext_vector_type(8)))  __bf16 v8bf;
typedef __attribute__((ext_vector_type(16))) __bf16 v16bf;
typedef __attribute__((ext_vector_type(8)))  float  v8f;

__device__ __forceinline__ v16bf cat8(v8bf lo, v8bf hi) {
  return __builtin_shufflevector(lo, hi, 0,1,2,3,4,5,6,7,8,9,10,11,12,13,14,15);
}
// A-matrix fragment (16x32 bf16): lane L holds row L%16, halves
// K = p..p+7 and K = p+16..p+23 with p = (L/16)*8  -> two 16B LDS loads.
__device__ __forceinline__ v16bf ld_fragA(const __bf16* p) {
  return cat8(*(const v8bf*)p, *(const v8bf*)(p + 16));
}
// B-matrix fragment (32x16 bf16): lane L holds col L%16, halves
// K = p..p+7 and K = p+16..p+23 with p = (L/16)*8  -> two 16B LDS loads.
__device__ __forceinline__ v16bf ld_fragB(const __bf16* p) {
  return cat8(*(const v8bf*)p, *(const v8bf*)(p + 16));
}
__device__ __forceinline__ v8f wmma_bf16(v16bf a, v16bf b, v8f c) {
  return __builtin_amdgcn_wmma_f32_16x16x32_bf16(false, a, false, b,
                                                 (short)0, c, false, false);
}

// 64x128x128 GEMM on LDS tiles. A: [64][LDP] bf16 (row stride LDP),
// B: [128][LDP] bf16 stored as [col][k] (k contiguous).
// Wave (rt = wave&3) covers rows rt*16..+15; (cg = wave>>2) cols cg*64..+63.
__device__ __forceinline__ void gemm64x128(const __bf16* A, const __bf16* B,
                                           int lane, int rt, int cg,
                                           v8f acc[4]) {
  const int l15 = lane & 15;
  const int lhi = lane >> 4;
  const int arow = rt * 16 + l15;
#pragma unroll
  for (int kk = 0; kk < DDIM; kk += 32) {
    v16bf a = ld_fragA(&A[arow * LDP + kk + lhi * 8]);
#pragma unroll
    for (int ct = 0; ct < 4; ++ct) {
      int bcol = cg * 64 + ct * 16 + l15;
      v16bf b = ld_fragB(&B[bcol * LDP + kk + lhi * 8]);
      acc[ct] = wmma_bf16(a, b, acc[ct]);
    }
  }
}

// ------------------------------ Phase 1 ------------------------------------
__global__ __launch_bounds__(256) void trimul_phase1(
    const float* __restrict__ x, const float* __restrict__ mask,
    const float* __restrict__ nw, const float* __restrict__ nb,
    const float* __restrict__ wl, const float* __restrict__ wr,
    const float* __restrict__ wlg, const float* __restrict__ wrg,
    const float* __restrict__ wog,
    __bf16* __restrict__ leftA, __bf16* __restrict__ rightR,
    __bf16* __restrict__ og) {
  __shared__ __bf16 xa[ROWS][LDP];      // normalized x, bf16      (17.4 KB)
  __shared__ __bf16 wbt[HDIM][LDP];     // weight / staging buffer (34.8 KB)

  const int tid = threadIdx.x;
  const int wave = tid >> 5;
  const int lane = tid & 31;
  const size_t m0 = (size_t)blockIdx.x * ROWS;     // 64 rows share one i
  const int iIdx = (int)(m0 / NDIM);
  const int k0 = (int)(m0 % NDIM);

  // ---- LayerNorm: 8 rows per wave, wave32 shuffle reduction ----
#pragma unroll
  for (int q = 0; q < 8; ++q) {
    int r = wave * 8 + q;
    const float* xr = x + (m0 + r) * DDIM;
    float4 v = ((const float4*)xr)[lane];
    float s = v.x + v.y + v.z + v.w;
    float ss = v.x * v.x + v.y * v.y + v.z * v.z + v.w * v.w;
#pragma unroll
    for (int off = 16; off > 0; off >>= 1) {
      s += __shfl_xor(s, off, 32);
      ss += __shfl_xor(ss, off, 32);
    }
    float mu = s * (1.0f / DDIM);
    float rstd = rsqrtf(ss * (1.0f / DDIM) - mu * mu + 1e-5f);
    int c = lane * 4;
    xa[r][c + 0] = (__bf16)((v.x - mu) * rstd * nw[c + 0] + nb[c + 0]);
    xa[r][c + 1] = (__bf16)((v.y - mu) * rstd * nw[c + 1] + nb[c + 1]);
    xa[r][c + 2] = (__bf16)((v.z - mu) * rstd * nw[c + 2] + nb[c + 2]);
    xa[r][c + 3] = (__bf16)((v.w - mu) * rstd * nw[c + 3] + nb[c + 3]);
  }
  __syncthreads();

  const int rt = wave & 3, cg = wave >> 2;
  const int l15 = lane & 15, lhi = lane >> 4;
  __bf16* cbh = (__bf16*)&wbt[0][0];    // staging alias [64][LDP] bf16

  const float* lins[2] = {wl, wr};
  const float* gats[2] = {wlg, wrg};
  __bf16* dsts[2] = {leftA, rightR};

  for (int p = 0; p < 2; ++p) {
    v8f accL[4] = {};
    v8f accG[4] = {};
    for (int idx = tid; idx < HDIM * DDIM; idx += 256)
      wbt[idx >> 7][idx & 127] = (__bf16)lins[p][idx];
    __syncthreads();
    gemm64x128(&xa[0][0], &wbt[0][0], lane, rt, cg, accL);
    __syncthreads();
    for (int idx = tid; idx < HDIM * DDIM; idx += 256)
      wbt[idx >> 7][idx & 127] = (__bf16)gats[p][idx];
    __syncthreads();
    gemm64x128(&xa[0][0], &wbt[0][0], lane, rt, cg, accG);
    __syncthreads();
    // combine: lin * mask * sigmoid(gate) -> bf16 staging (C layout:
    // VGPR rr -> row rt*16 + (lane/16)*8 + rr, col = cg*64 + ct*16 + lane%16)
#pragma unroll
    for (int ct = 0; ct < 4; ++ct) {
      int col = cg * 64 + ct * 16 + l15;
#pragma unroll
      for (int rr = 0; rr < 8; ++rr) {
        int row = rt * 16 + lhi * 8 + rr;
        float mval = mask[m0 + row];
        float g = 1.0f / (1.0f + __expf(-accG[ct][rr]));
        cbh[row * LDP + col] = (__bf16)(accL[ct][rr] * mval * g);
      }
    }
    __syncthreads();
    // transpose-store: dst[d][iIdx][k0 + 0..63], 64B contiguous per thread
    {
      int d = tid >> 1, off = (tid & 1) * 32;
      __bf16 tmp[32] __attribute__((aligned(16)));
#pragma unroll
      for (int r = 0; r < 32; ++r) tmp[r] = cbh[(off + r) * LDP + d];
      __bf16* gp = dsts[p] + ((size_t)d * NDIM + iIdx) * NDIM + k0 + off;
#pragma unroll
      for (int v = 0; v < 4; ++v) ((uint4*)gp)[v] = ((const uint4*)tmp)[v];
    }
    __syncthreads();
  }

  // ---- out gate: sigmoid only, stored row-major [m][d] bf16 ----
  {
    v8f accG[4] = {};
    for (int idx = tid; idx < HDIM * DDIM; idx += 256)
      wbt[idx >> 7][idx & 127] = (__bf16)wog[idx];
    __syncthreads();
    gemm64x128(&xa[0][0], &wbt[0][0], lane, rt, cg, accG);
    __syncthreads();
#pragma unroll
    for (int ct = 0; ct < 4; ++ct) {
      int col = cg * 64 + ct * 16 + l15;
#pragma unroll
      for (int rr = 0; rr < 8; ++rr) {
        int row = rt * 16 + lhi * 8 + rr;
        cbh[row * LDP + col] = (__bf16)(1.0f / (1.0f + __expf(-accG[ct][rr])));
      }
    }
    __syncthreads();
    int row = tid >> 2, off = (tid & 3) * 32;
    __bf16 tmp[32] __attribute__((aligned(16)));
#pragma unroll
    for (int r = 0; r < 32; ++r) tmp[r] = cbh[row * LDP + off + r];
    __bf16* gp = og + (m0 + row) * HDIM + off;
#pragma unroll
    for (int v = 0; v < 4; ++v) ((uint4*)gp)[v] = ((const uint4*)tmp)[v];
  }
}

// ------------------------------ Phase 2 ------------------------------------
// grid (N/128, N/128, H); one 128x128 output tile per block per channel.
// d as blockIdx.z groups same-channel tiles in time -> A_d/R_d (2.4 MB) live
// in the 192 MB L2 while their 36 tiles execute.
__global__ __launch_bounds__(256) void trimul_phase2(
    const __bf16* __restrict__ leftA, const __bf16* __restrict__ rightR,
    float* __restrict__ out_mid) {
  __shared__ __bf16 As[2][128][40];     // [i][k] k-contig, padded (20.5 KB)
  __shared__ __bf16 Bs[2][128][40];     // [j][k] k-contig, padded (20.5 KB)

  const int tid = threadIdx.x;
  const int wave = tid >> 5;
  const int lane = tid & 31;
  const int i0 = blockIdx.y * 128, j0 = blockIdx.x * 128, d = blockIdx.z;
  const __bf16* Ad = leftA + (size_t)d * NDIM * NDIM;
  const __bf16* Rd = rightR + (size_t)d * NDIM * NDIM;
  const int lr = tid >> 1, lo = (tid & 1) * 16;

  auto load_stage = [&](int s, int kk) {
    const __bf16* ga = Ad + (size_t)(i0 + lr) * NDIM + kk + lo;
    const __bf16* gb = Rd + (size_t)(j0 + lr) * NDIM + kk + lo;
#if USE_ASYNC_LDS
    // Async DMA global->LDS (16B per lane per op), tracked by ASYNCcnt.
    __builtin_amdgcn_global_load_async_to_lds_b128(
        (gptr_t)ga, (lptr_t)&As[s][lr][lo], 0, 0);
    __builtin_amdgcn_global_load_async_to_lds_b128(
        (gptr_t)(ga + 8), (lptr_t)&As[s][lr][lo + 8], 0, 0);
    __builtin_amdgcn_global_load_async_to_lds_b128(
        (gptr_t)gb, (lptr_t)&Bs[s][lr][lo], 0, 0);
    __builtin_amdgcn_global_load_async_to_lds_b128(
        (gptr_t)(gb + 8), (lptr_t)&Bs[s][lr][lo + 8], 0, 0);
#else
    *(uint4*)&As[s][lr][lo]     = ((const uint4*)ga)[0];
    *(uint4*)&As[s][lr][lo + 8] = ((const uint4*)ga)[1];
    *(uint4*)&Bs[s][lr][lo]     = ((const uint4*)gb)[0];
    *(uint4*)&Bs[s][lr][lo + 8] = ((const uint4*)gb)[1];
#endif
  };
  auto async_join = [&]() {
#if USE_ASYNC_LDS
    __builtin_amdgcn_s_wait_asynccnt(0);   // s_wait_asynccnt 0
#endif
  };

  load_stage(0, 0);
  async_join();
  __syncthreads();

  const int wr = wave & 3, wc = wave >> 2;   // rows wr*32(+16), cols wc*64
  const int l15 = lane & 15, lhi = lane >> 4;
  v8f acc[2][4] = {};

  for (int kt = 0; kt < NDIM / 32; ++kt) {
    int s = kt & 1;
    if (kt + 1 < NDIM / 32) load_stage(s ^ 1, (kt + 1) * 32);
    v16bf a0 = ld_fragA(&As[s][wr * 32      + l15][lhi * 8]);
    v16bf a1 = ld_fragA(&As[s][wr * 32 + 16 + l15][lhi * 8]);
#pragma unroll
    for (int ct = 0; ct < 4; ++ct) {
      v16bf b = ld_fragB(&Bs[s][wc * 64 + ct * 16 + l15][lhi * 8]);
      acc[0][ct] = wmma_bf16(a0, b, acc[0][ct]);
      acc[1][ct] = wmma_bf16(a1, b, acc[1][ct]);
    }
    async_join();      // next stage fully in LDS before the barrier releases
    __syncthreads();
  }

  // out_mid[d][i][j]: 16-lane segments are j-contiguous (64B)
#pragma unroll
  for (int h = 0; h < 2; ++h) {
#pragma unroll
    for (int ct = 0; ct < 4; ++ct) {
      int row0 = i0 + wr * 32 + h * 16 + lhi * 8;
      int col = j0 + wc * 64 + ct * 16 + l15;
      float* gp = out_mid + ((size_t)d * NDIM + row0) * NDIM + col;
#pragma unroll
      for (int rr = 0; rr < 8; ++rr) gp[(size_t)rr * NDIM] = acc[h][ct][rr];
    }
  }
}

// ------------------------------ Phase 3 ------------------------------------
__global__ __launch_bounds__(256) void trimul_phase3(
    const float* __restrict__ out_mid, const __bf16* __restrict__ og,
    const float* __restrict__ onw, const float* __restrict__ onb,
    const float* __restrict__ wout, float* __restrict__ out) {
  __shared__ __attribute__((aligned(16))) char region[HDIM * 68 * 4]; // 34.8KB
  __shared__ __bf16 gm[ROWS][LDP];                                    // 17.4KB
  float* sm = (float*)region;           // [128][68]  gathered out_mid
  __bf16* wob = (__bf16*)region;        // [128][LDP] w_out (reuses sm)
  float* cfs = (float*)region;          // [64][132]  C staging (reuses wob)

  const int tid = threadIdx.x;
  const int wave = tid >> 5;
  const int lane = tid & 31;
  const size_t m0 = (size_t)blockIdx.x * ROWS;
  const int iIdx = (int)(m0 / NDIM);
  const int j0 = (int)(m0 % NDIM);

  // gather sm[d][j] <- out_mid[d][iIdx][j0..j0+63] (128B contig per thread)
  {
    int dd = tid >> 1, off = (tid & 1) * 32;
    const float* gp = out_mid + ((size_t)dd * NDIM + iIdx) * NDIM + j0 + off;
#pragma unroll
    for (int v = 0; v < 8; ++v)
      ((float4*)&sm[dd * 68 + off])[v] = ((const float4*)gp)[v];
  }
  __syncthreads();

  // LayerNorm over d (+ sigmoid gate) -> gm[j][h] bf16
  if (tid < ROWS) {
    int j = tid;
    float s = 0.f, ss = 0.f;
    for (int d = 0; d < HDIM; ++d) {
      float v = sm[d * 68 + j];
      s += v;
      ss += v * v;
    }
    float mu = s * (1.0f / HDIM);
    float rstd = rsqrtf(ss * (1.0f / HDIM) - mu * mu + 1e-5f);
    const __bf16* ogr = og + (m0 + j) * HDIM;
    for (int d = 0; d < HDIM; ++d) {
      float v = (sm[d * 68 + j] - mu) * rstd * onw[d] + onb[d];
      gm[j][d] = (__bf16)(v * (float)ogr[d]);
    }
  }
  __syncthreads();

  // w_out [dout][h] copied bf16 (h contiguous -> B-fragment friendly)
  for (int idx = tid; idx < DDIM * HDIM; idx += 256)
    wob[(idx >> 7) * LDP + (idx & 127)] = (__bf16)wout[idx];
  __syncthreads();

  const int rt = wave & 3, cg = wave >> 2;
  const int l15 = lane & 15, lhi = lane >> 4;
  v8f acc[4] = {};
  gemm64x128(&gm[0][0], wob, lane, rt, cg, acc);
  __syncthreads();

#pragma unroll
  for (int ct = 0; ct < 4; ++ct) {
    int col = cg * 64 + ct * 16 + l15;
#pragma unroll
    for (int rr = 0; rr < 8; ++rr) {
      int row = rt * 16 + lhi * 8 + rr;
      cfs[row * 132 + col] = acc[ct][rr];
    }
  }
  __syncthreads();

  // coalesced fp32 output: out[(m0+row)][0..127]
  {
    int row = tid >> 2, off = (tid & 3) * 32;
    float* gp = out + (m0 + row) * DDIM + off;
#pragma unroll
    for (int v = 0; v < 8; ++v)
      ((float4*)gp)[v] = ((const float4*)&cfs[row * 132 + off])[v];
  }
}

// ------------------------------ Launch -------------------------------------
extern "C" void kernel_launch(void* const* d_in, const int* in_sizes, int n_in,
                              void* d_out, int out_size, void* d_ws,
                              size_t ws_size, hipStream_t stream) {
  (void)in_sizes; (void)n_in; (void)out_size; (void)ws_size;
  const float* x    = (const float*)d_in[0];
  const float* mask = (const float*)d_in[1];
  const float* nw   = (const float*)d_in[2];
  const float* nb   = (const float*)d_in[3];
  const float* wl   = (const float*)d_in[4];
  const float* wr   = (const float*)d_in[5];
  const float* wlg  = (const float*)d_in[6];
  const float* wrg  = (const float*)d_in[7];
  const float* wog  = (const float*)d_in[8];
  const float* onw  = (const float*)d_in[9];
  const float* onb  = (const float*)d_in[10];
  const float* wout = (const float*)d_in[11];

  const size_t SZ = (size_t)HDIM * NDIM * NDIM;   // per-tensor element count
  __bf16* leftA  = (__bf16*)d_ws;
  __bf16* rightR = leftA + SZ;
  __bf16* og     = rightR + SZ;
  float*  outmid = (float*)(og + SZ);

  const int nblk = (NDIM * NDIM) / ROWS;          // 9216
  dim3 blk(256);

  trimul_phase1<<<nblk, blk, 0, stream>>>(x, mask, nw, nb, wl, wr, wlg, wrg,
                                          wog, leftA, rightR, og);
  trimul_phase2<<<dim3(NDIM / 128, NDIM / 128, HDIM), blk, 0, stream>>>(
      leftA, rightR, outmid);
  trimul_phase3<<<nblk, blk, 0, stream>>>(outmid, og, onw, onb, wout,
                                          (float*)d_out);
}
